// MDCT_19722489823668
// MI455X (gfx1250) — compile-verified, matrix operand order checked by reading, not run on hardware
//
#include <hip/hip_runtime.h>
#include <hip/hip_bf16.h>
#include <math.h>

typedef __attribute__((ext_vector_type(16))) _Float16 v16h;
typedef __attribute__((ext_vector_type(8)))  _Float16 v8h;
typedef __attribute__((ext_vector_type(8)))  float    v8f;

// Problem constants (from reference)
constexpr int kBatch   = 16;
constexpr int kTAudio  = 512000;
constexpr int kFrame   = 1024;   // K of the GEMM
constexpr int kHop     = 512;
constexpr int kNBins   = 512;    // N of the GEMM
constexpr int kL       = 1000;   // frames per batch
constexpr int kMRows   = kBatch * kL;  // 16000
constexpr int kKSteps  = kFrame / 32;  // 32 K-steps of K=32

// ---------------------------------------------------------------------------
// Kernel 1: gather zero-padded overlapping frames, convert to f16.
// Xf[r, k] = audio[b, l*512 + k - 256] (0 outside), r = b*1000 + l.
// ---------------------------------------------------------------------------
__global__ __launch_bounds__(256)
void mdct_frames_f16(const float* __restrict__ audio, _Float16* __restrict__ Xf) {
    const int chunksPerRow = kFrame / 8;                  // 128
    int idx = blockIdx.x * blockDim.x + threadIdx.x;      // one 8-elem chunk
    if (idx >= kMRows * chunksPerRow) return;
    int r  = idx / chunksPerRow;
    int kc = (idx - r * chunksPerRow) * 8;
    int b  = r / kL;
    int l  = r - b * kL;
    long base = (long)l * kHop - 256 + kc;                // multiple of 8 -> 32B aligned
    const float* src = audio + (long)b * kTAudio;

    _Float16 tmp[8];
    if (base >= 0 && base + 8 <= kTAudio) {
        const float4* p = (const float4*)(src + base);
        float4 f0 = p[0], f1 = p[1];
        tmp[0] = (_Float16)f0.x; tmp[1] = (_Float16)f0.y;
        tmp[2] = (_Float16)f0.z; tmp[3] = (_Float16)f0.w;
        tmp[4] = (_Float16)f1.x; tmp[5] = (_Float16)f1.y;
        tmp[6] = (_Float16)f1.z; tmp[7] = (_Float16)f1.w;
    } else {
        #pragma unroll
        for (int j = 0; j < 8; ++j) {
            long s = base + j;
            float v = (s >= 0 && s < kTAudio) ? src[s] : 0.0f;
            tmp[j] = (_Float16)v;
        }
    }
    *(v8h*)(Xf + (long)r * kFrame + kc) = *(const v8h*)tmp;
}

// ---------------------------------------------------------------------------
// Kernel 2: MDCT basis, transposed, window + 1/16 scale folded in.
// Bt[n, k] = (1/16) * sin(pi*(k+0.5)/1024) * cos(pi*(k+256.5)*(2n+1)/1024)
// ---------------------------------------------------------------------------
__global__ __launch_bounds__(256)
void mdct_basis_f16(_Float16* __restrict__ Bt) {
    int idx = blockIdx.x * blockDim.x + threadIdx.x;
    if (idx >= kNBins * kFrame) return;
    int n = idx / kFrame;
    int k = idx - n * kFrame;
    double win = sin(M_PI * ((double)k + 0.5) / (double)kFrame);
    double ang = M_PI * (double)(2 * n + 1) * ((double)k + 256.5) / (double)kFrame;
    double v   = (1.0 / 16.0) * win * cos(ang);
    Bt[idx] = (_Float16)(float)v;
}

// ---------------------------------------------------------------------------
// Kernel 3: WMMA GEMM with software pipelining.
// Out[16000 x 512] = Xf[16000 x 1024] * Bt^T, Bt stored N-major [512 x 1024].
//
// Block = 256 threads = 8 waves; block owns a 32-row stripe, wave w owns
// columns [w*64, w*64+64). Per K-step (K=32): 2 A fragments x 4 B fragments
// -> 8 WMMAs. Ping/pong fragment buffers; the manual unroll-by-2 is the only
// unrolling (compiler re-unroll disabled) so accumulators keep fixed
// registers and no live-range copies are needed.
// ---------------------------------------------------------------------------
union Frag { v16h v; struct { v8h lo; v8h hi; } p; };

struct FragSet {
    Frag a0, b0, a1, b1, b2, b3;   // load order: first WMMA's operands first
};

__device__ __forceinline__
void load_fragset(FragSet& f, const _Float16* arow0, const _Float16* brow,
                  int step, int halfsel) {
    const int off = step * 32 + (halfsel ? 8 : 0);
    f.a0.p.lo = *(const v8h*)(arow0 + off);
    f.a0.p.hi = *(const v8h*)(arow0 + off + 16);
    f.b0.p.lo = *(const v8h*)(brow + off);
    f.b0.p.hi = *(const v8h*)(brow + off + 16);
    f.a1.p.lo = *(const v8h*)(arow0 + 16 * kFrame + off);
    f.a1.p.hi = *(const v8h*)(arow0 + 16 * kFrame + off + 16);
    f.b1.p.lo = *(const v8h*)(brow + 16 * kFrame + off);
    f.b1.p.hi = *(const v8h*)(brow + 16 * kFrame + off + 16);
    f.b2.p.lo = *(const v8h*)(brow + 32 * kFrame + off);
    f.b2.p.hi = *(const v8h*)(brow + 32 * kFrame + off + 16);
    f.b3.p.lo = *(const v8h*)(brow + 48 * kFrame + off);
    f.b3.p.hi = *(const v8h*)(brow + 48 * kFrame + off + 16);
}

// Two WMMAs per B fragment (shortest B live ranges).
#define MDCT_COMPUTE_SET(F)                                                     \
    do {                                                                        \
        acc0 = __builtin_amdgcn_wmma_f32_16x16x32_f16(false, (F).a0.v, false,   \
                   (F).b0.v, (short)0, acc0, false, false);                     \
        acc4 = __builtin_amdgcn_wmma_f32_16x16x32_f16(false, (F).a1.v, false,   \
                   (F).b0.v, (short)0, acc4, false, false);                     \
        acc1 = __builtin_amdgcn_wmma_f32_16x16x32_f16(false, (F).a0.v, false,   \
                   (F).b1.v, (short)0, acc1, false, false);                     \
        acc5 = __builtin_amdgcn_wmma_f32_16x16x32_f16(false, (F).a1.v, false,   \
                   (F).b1.v, (short)0, acc5, false, false);                     \
        acc2 = __builtin_amdgcn_wmma_f32_16x16x32_f16(false, (F).a0.v, false,   \
                   (F).b2.v, (short)0, acc2, false, false);                     \
        acc6 = __builtin_amdgcn_wmma_f32_16x16x32_f16(false, (F).a1.v, false,   \
                   (F).b2.v, (short)0, acc6, false, false);                     \
        acc3 = __builtin_amdgcn_wmma_f32_16x16x32_f16(false, (F).a0.v, false,   \
                   (F).b3.v, (short)0, acc3, false, false);                     \
        acc7 = __builtin_amdgcn_wmma_f32_16x16x32_f16(false, (F).a1.v, false,   \
                   (F).b3.v, (short)0, acc7, false, false);                     \
    } while (0)

__global__ __launch_bounds__(256)
void mdct_wmma_gemm(const _Float16* __restrict__ Xf,
                    const _Float16* __restrict__ Bt,
                    float* __restrict__ Out) {
    const int lane    = threadIdx.x & 31;
    const int wave    = threadIdx.x >> 5;        // 0..7
    const int lid     = lane & 15;
    const int halfsel = (lane >> 4) & 1;

    const int mbase = blockIdx.x * 32;
    const int nbase = wave * 64;

    const _Float16* arow0 = Xf + (long)(mbase + lid) * kFrame;
    const _Float16* brow  = Bt + (long)(nbase + lid) * kFrame;

    // Warm the cache for this wave's four B rows (global_prefetch_b8).
    __builtin_prefetch(brow, 0, 3);
    __builtin_prefetch(brow + 16 * kFrame, 0, 3);
    __builtin_prefetch(brow + 32 * kFrame, 0, 3);
    __builtin_prefetch(brow + 48 * kFrame, 0, 3);

    v8f acc0 = {}, acc1 = {}, acc2 = {}, acc3 = {};
    v8f acc4 = {}, acc5 = {}, acc6 = {}, acc7 = {};

    FragSet P, Q;
    load_fragset(P, arow0, brow, 0, halfsel);
    #pragma clang loop unroll(disable)
    for (int s = 1; s < kKSteps - 1; s += 2) {
        load_fragset(Q, arow0, brow, s, halfsel);       // loads for step s
        MDCT_COMPUTE_SET(P);                            // WMMAs for step s-1
        load_fragset(P, arow0, brow, s + 1, halfsel);   // loads for step s+1
        MDCT_COMPUTE_SET(Q);                            // WMMAs for step s
    }
    load_fragset(Q, arow0, brow, kKSteps - 1, halfsel); // step 31 loads
    MDCT_COMPUTE_SET(P);                                // step 30 WMMAs
    MDCT_COMPUTE_SET(Q);                                // step 31 WMMAs

    // C/D layout: VGPR v -> row M=v (lanes 0-15) or M=v+8 (lanes 16-31);
    // N = lane % 16. Stores coalesce across lid.
    const int rowoff = halfsel * 8;
    #pragma unroll
    for (int v = 0; v < 8; ++v) {
        float* o0 = Out + (long)(mbase + v + rowoff) * kNBins + nbase + lid;
        o0[0]  = acc0[v];
        o0[16] = acc1[v];
        o0[32] = acc2[v];
        o0[48] = acc3[v];
        float* o1 = o0 + 16 * kNBins;   // rows +16 for the a1 tile set
        o1[0]  = acc4[v];
        o1[16] = acc5[v];
        o1[32] = acc6[v];
        o1[48] = acc7[v];
    }
}

// ---------------------------------------------------------------------------
extern "C" void kernel_launch(void* const* d_in, const int* in_sizes, int n_in,
                              void* d_out, int out_size, void* d_ws, size_t ws_size,
                              hipStream_t stream) {
    const float* audio = (const float*)d_in[0];
    float* out = (float*)d_out;

    // Workspace: Xf [16000 x 1024] f16 (32.77 MB), then Bt [512 x 1024] f16 (1.05 MB)
    _Float16* Xf = (_Float16*)d_ws;
    _Float16* Bt = (_Float16*)((char*)d_ws + (size_t)kMRows * kFrame * sizeof(_Float16));

    const int chunks = kMRows * (kFrame / 8);
    mdct_frames_f16<<<(chunks + 255) / 256, 256, 0, stream>>>(audio, Xf);

    const int belems = kNBins * kFrame;
    mdct_basis_f16<<<(belems + 255) / 256, 256, 0, stream>>>(Bt);

    mdct_wmma_gemm<<<kMRows / 32, 256, 0, stream>>>(Xf, Bt, out);
}
